// MultiHeadAttention_17386027615012
// MI455X (gfx1250) — compile-verified, matrix operand order checked by reading, not run on hardware
//
#include <hip/hip_runtime.h>
#include <hip/hip_bf16.h>

// ---------------------------------------------------------------------------
// MultiHeadAttention (B=2, N=4096, EMB=768, HEADS=8, HEAD_DIM=96)
// softmax over the HEAD axis (replicating reference), / sqrt(768) after.
// All GEMMs on v_wmma_f32_16x16x32_bf16 (CDNA5 wave32 WMMA), f32 accumulate.
// Pre-passes convert x / weights to bf16 (weights transposed to N-major) so
// every WMMA fragment load is two contiguous 16-byte loads.
// ---------------------------------------------------------------------------

#define EMB   768
#define HEADS 8
#define HD    96
#define BATCH 2
#define SEQ   4096
#define QKV3  (3 * EMB)

typedef __attribute__((ext_vector_type(16))) __bf16 v16bf;
typedef __attribute__((ext_vector_type(8)))  float  v8f;

__device__ __forceinline__ __bf16 f2bf(float x) {
  return static_cast<__bf16>(x);   // native v_cvt on gfx1250
}

__device__ __forceinline__ v8f wmma_bf16(v16bf a, v16bf b, v8f c) {
  // (neg_a, A, neg_b, B, c_mod, C, reuse_a, reuse_b)
  return __builtin_amdgcn_wmma_f32_16x16x32_bf16(false, a, false, b,
                                                 (short)0, c, false, false);
}

// A/B fragment (16x32 / 32x16) from a bf16 matrix whose WMMA "row" dimension
// (A: M-row, B: N-column) is the slow axis and the K dimension is contiguous.
// ISA 7.12.2 layout: lane<16 -> row=lane, K={0..7,16..23}; lane>=16 -> +8.
__device__ __forceinline__ v16bf frag_bf16(const __bf16* __restrict__ p0,
                                           int ld, int lane) {
  const __bf16* p = p0 + (size_t)(lane & 15) * ld + ((lane >> 4) & 1) * 8;
  v16bf a;
#pragma unroll
  for (int i = 0; i < 8; ++i) a[i] = p[i];           // 16B contiguous
#pragma unroll
  for (int i = 0; i < 8; ++i) a[i + 8] = p[16 + i];  // 16B contiguous
  return a;
}

// Same fragment gather from an LDS bf16 tile (generic address space).
__device__ __forceinline__ v16bf frag_lds_bf16(const __bf16* p0, int ld,
                                               int lane) {
  const __bf16* p = p0 + (lane & 15) * ld + ((lane >> 4) & 1) * 8;
  v16bf a;
#pragma unroll
  for (int i = 0; i < 8; ++i) a[i] = p[i];           // ds_load 16B
#pragma unroll
  for (int i = 0; i < 8; ++i) a[i + 8] = p[16 + i];  // ds_load 16B
  return a;
}

// ---------------------------------------------------------------------------
// Pre-pass A: contiguous f32 -> bf16 (x)
// ---------------------------------------------------------------------------
__global__ __launch_bounds__(256) void cvt_bf16_kernel(
    const float* __restrict__ in, __bf16* __restrict__ out, int n) {
  const int i = (blockIdx.x * 256 + threadIdx.x) * 4;
  if (i < n) {
    const float4 v = *(const float4*)(in + i);
    out[i + 0] = f2bf(v.x);
    out[i + 1] = f2bf(v.y);
    out[i + 2] = f2bf(v.z);
    out[i + 3] = f2bf(v.w);
  }
}

// ---------------------------------------------------------------------------
// Pre-pass B: transpose + convert weights: W[K][N] f32 -> Wt[N][K] bf16.
// ---------------------------------------------------------------------------
__global__ __launch_bounds__(256) void transpose_cvt_kernel(
    const float* __restrict__ w, __bf16* __restrict__ wt, int K, int N) {
  __shared__ __bf16 tile[32][33];
  const int n0 = blockIdx.x * 32;
  const int k0 = blockIdx.y * 32;
  const int tx = threadIdx.x & 31;
  const int ty = threadIdx.x >> 5;  // 0..7
#pragma unroll
  for (int i = 0; i < 4; ++i) {
    const int k = ty + i * 8;
    tile[k][tx] = f2bf(w[(size_t)(k0 + k) * N + n0 + tx]);
  }
  __syncthreads();
#pragma unroll
  for (int i = 0; i < 4; ++i) {
    const int n = ty + i * 8;
    wt[(size_t)(n0 + n) * K + k0 + tx] = tile[tx][n];
  }
}

// ---------------------------------------------------------------------------
// Kernel 1: qkv = xb @ Wqt^T + b, split into head-major bf16 Q/K and
// d-major (transposed) V.  WG = 8 waves (4x2), 128x128 tile, 32x64 per wave.
// ---------------------------------------------------------------------------
__global__ __launch_bounds__(256) void qkv_kernel(
    const __bf16* __restrict__ xb, const __bf16* __restrict__ wt,
    const float* __restrict__ bias, __bf16* __restrict__ Qb,
    __bf16* __restrict__ Kb, __bf16* __restrict__ Vt) {
  const int lane = threadIdx.x & 31;
  const int wave = threadIdx.x >> 5;
  const int m0 = blockIdx.x * 128 + (wave >> 1) * 32;  // row in [0, B*N)
  const int n0 = blockIdx.y * 128 + (wave & 1) * 64;   // col in [0, 2304)

  v8f acc[2][4] = {};
#pragma unroll 2
  for (int k = 0; k < EMB; k += 32) {
    v16bf a0 = frag_bf16(xb + (size_t)m0 * EMB + k, EMB, lane);
    v16bf a1 = frag_bf16(xb + (size_t)(m0 + 16) * EMB + k, EMB, lane);
#pragma unroll
    for (int ni = 0; ni < 4; ++ni) {
      v16bf bf = frag_bf16(wt + (size_t)(n0 + ni * 16) * EMB + k, EMB, lane);
      acc[0][ni] = wmma_bf16(a0, bf, acc[0][ni]);
      acc[1][ni] = wmma_bf16(a1, bf, acc[1][ni]);
    }
  }

  const int rbase = ((lane >> 4) & 1) * 8;
  const int cl = lane & 15;
#pragma unroll
  for (int ni = 0; ni < 4; ++ni) {
    const int col = n0 + ni * 16 + cl;  // (h, d, s) interleaved
    const int h = col / (3 * HD);
    const int rem = col % (3 * HD);
    const int d = rem / 3;
    const int s = rem % 3;
    const float bv = bias[col];
#pragma unroll
    for (int qi = 0; qi < 2; ++qi)
#pragma unroll
      for (int v = 0; v < 8; ++v) {
        const int row = m0 + qi * 16 + rbase + v;  // b*SEQ + n
        const int b = row >> 12;
        const int n = row & (SEQ - 1);
        const __bf16 val = f2bf(acc[qi][ni][v] + bv);
        if (s == 0)
          Qb[((((size_t)b * HEADS + h) * SEQ) + n) * HD + d] = val;
        else if (s == 1)
          Kb[((((size_t)b * HEADS + h) * SEQ) + n) * HD + d] = val;
        else  // V stored d-major per head for contiguous PV B-fragments
          Vt[((((size_t)b * HEADS + h) * HD) + d) * SEQ + n] = val;
      }
  }
}

// ---------------------------------------------------------------------------
// Kernel 2: attention with softmax over the HEAD axis.
// One workgroup = 8 waves = 8 heads, one 32-row q-tile, streams 32-wide
// k-tiles.  exp(S) goes to LDS; the cross-head reduction pass directly emits
// the normalized bf16 P tiles for all heads (no duplicated per-wave work).
// ---------------------------------------------------------------------------
__global__ __launch_bounds__(256) void attn_kernel(
    const __bf16* __restrict__ Qb, const __bf16* __restrict__ Kb,
    const __bf16* __restrict__ Vt, __bf16* __restrict__ Ob) {
  __shared__ float  expS[HEADS][32][32];  // 32 KB
  __shared__ __bf16 Pb[HEADS][32][32];    // 16 KB: normalized att, bf16

  const int lane = threadIdx.x & 31;
  const int h = threadIdx.x >> 5;  // wave == head
  const int b = blockIdx.y;
  const int q0 = blockIdx.x * 32;

  const __bf16* Qh = Qb + (((size_t)b * HEADS + h) * SEQ) * HD;
  const __bf16* Kh = Kb + (((size_t)b * HEADS + h) * SEQ) * HD;
  const __bf16* Vh = Vt + (((size_t)b * HEADS + h) * HD) * SEQ;  // [d][n]

  // Preload Q fragments for the whole k-loop: 2 q-subtiles x 3 d-steps.
  v16bf qf[2][3];
#pragma unroll
  for (int qi = 0; qi < 2; ++qi)
#pragma unroll
    for (int j = 0; j < 3; ++j)
      qf[qi][j] =
          frag_bf16(Qh + (size_t)(q0 + qi * 16) * HD + j * 32, HD, lane);

  v8f o[2][6] = {};
  const float inv_scale = 0.03608439182435161f;  // 1/sqrt(768)
  const int rbase = ((lane >> 4) & 1) * 8;
  const int cl = lane & 15;

  for (int kt = 0; kt < SEQ; kt += 32) {
    // prefetch next k-tile of K / V (global_prefetch_b8)
    if (kt + 32 < SEQ) {
      __builtin_prefetch(Kh + (size_t)(kt + 32) * HD, 0, 3);
      __builtin_prefetch(Vh + (size_t)(kt + 32), 0, 3);
    }

    // ---- S = Q K^T for this head ----
    v8f s[2][2] = {};
#pragma unroll
    for (int j = 0; j < 3; ++j) {
      v16bf kf0 = frag_bf16(Kh + (size_t)(kt + 0) * HD + j * 32, HD, lane);
      v16bf kf1 = frag_bf16(Kh + (size_t)(kt + 16) * HD + j * 32, HD, lane);
      s[0][0] = wmma_bf16(qf[0][j], kf0, s[0][0]);
      s[0][1] = wmma_bf16(qf[0][j], kf1, s[0][1]);
      s[1][0] = wmma_bf16(qf[1][j], kf0, s[1][0]);
      s[1][1] = wmma_bf16(qf[1][j], kf1, s[1][1]);
    }

    // ---- exp(S) -> LDS ----
#pragma unroll
    for (int qi = 0; qi < 2; ++qi)
#pragma unroll
      for (int ki = 0; ki < 2; ++ki)
#pragma unroll
        for (int v = 0; v < 8; ++v)
          expS[h][qi * 16 + rbase + v][ki * 16 + cl] = __expf(s[qi][ki][v]);
    __syncthreads();

    // ---- cross-head softmax: reduce over heads and pack bf16 P tiles ----
    // 1024 (q,k) cells over 256 threads; each thread owns 4 cells and emits
    // the normalized value for all 8 heads (no per-wave duplication).
#pragma unroll
    for (int e = threadIdx.x; e < 1024; e += 256) {
      const int qq = e >> 5, kk = e & 31;
      float ev[HEADS];
      float t = 0.f;
#pragma unroll
      for (int h2 = 0; h2 < HEADS; ++h2) {
        ev[h2] = expS[h2][qq][kk];
        t += ev[h2];
      }
      const float r = inv_scale / t;
#pragma unroll
      for (int h2 = 0; h2 < HEADS; ++h2)
        Pb[h2][qq][kk] = f2bf(ev[h2] * r);
    }
    __syncthreads();

    // ---- O += P @ V  (P from LDS as 16B ds loads; V d-major global) ----
    v16bf pf0 = frag_lds_bf16(&Pb[h][0][0], 32, lane);
    v16bf pf1 = frag_lds_bf16(&Pb[h][16][0], 32, lane);
#pragma unroll
    for (int dj = 0; dj < 6; ++dj) {
      v16bf vf = frag_bf16(Vh + (size_t)(dj * 16) * SEQ + kt, SEQ, lane);
      o[0][dj] = wmma_bf16(pf0, vf, o[0][dj]);
      o[1][dj] = wmma_bf16(pf1, vf, o[1][dj]);
    }
    __syncthreads();  // expS/Pb reuse next iteration
  }

  // ---- write O in [b][n][(h d)] bf16 layout for the output projection ----
#pragma unroll
  for (int qi = 0; qi < 2; ++qi)
#pragma unroll
    for (int dj = 0; dj < 6; ++dj)
#pragma unroll
      for (int v = 0; v < 8; ++v) {
        const int n = q0 + qi * 16 + rbase + v;
        const int d = dj * 16 + cl;
        Ob[((size_t)(b * SEQ + n) * EMB) + h * HD + d] = f2bf(o[qi][dj][v]);
      }
}

// ---------------------------------------------------------------------------
// Kernel 3: out = Ob @ Wpt^T + b   ([8192x768] x [768x768] -> f32)
// ---------------------------------------------------------------------------
__global__ __launch_bounds__(256) void proj_kernel(
    const __bf16* __restrict__ Ob, const __bf16* __restrict__ wt,
    const float* __restrict__ bias, float* __restrict__ out) {
  const int lane = threadIdx.x & 31;
  const int wave = threadIdx.x >> 5;
  const int m0 = blockIdx.x * 128 + (wave >> 1) * 32;
  const int n0 = blockIdx.y * 128 + (wave & 1) * 64;

  v8f acc[2][4] = {};
#pragma unroll 2
  for (int k = 0; k < EMB; k += 32) {
    v16bf a0 = frag_bf16(Ob + (size_t)m0 * EMB + k, EMB, lane);
    v16bf a1 = frag_bf16(Ob + (size_t)(m0 + 16) * EMB + k, EMB, lane);
#pragma unroll
    for (int ni = 0; ni < 4; ++ni) {
      v16bf bf = frag_bf16(wt + (size_t)(n0 + ni * 16) * EMB + k, EMB, lane);
      acc[0][ni] = wmma_bf16(a0, bf, acc[0][ni]);
      acc[1][ni] = wmma_bf16(a1, bf, acc[1][ni]);
    }
  }

  const int rbase = ((lane >> 4) & 1) * 8;
  const int cl = lane & 15;
#pragma unroll
  for (int ni = 0; ni < 4; ++ni) {
    const int col = n0 + ni * 16 + cl;
    const float bv = bias[col];
#pragma unroll
    for (int qi = 0; qi < 2; ++qi)
#pragma unroll
      for (int v = 0; v < 8; ++v) {
        const int row = m0 + qi * 16 + rbase + v;
        out[(size_t)row * EMB + col] = acc[qi][ni][v] + bv;
      }
  }
}

// ---------------------------------------------------------------------------
extern "C" void kernel_launch(void* const* d_in, const int* in_sizes, int n_in,
                              void* d_out, int out_size, void* d_ws,
                              size_t ws_size, hipStream_t stream) {
  const float* x      = (const float*)d_in[0];
  const float* qkv_w  = (const float*)d_in[1];
  const float* qkv_b  = (const float*)d_in[2];
  const float* proj_w = (const float*)d_in[3];
  const float* proj_b = (const float*)d_in[4];
  float* out = (float*)d_out;

  const size_t nX   = (size_t)BATCH * SEQ * EMB;       // 6,291,456
  const size_t nWq  = (size_t)EMB * QKV3;              // 1,769,472
  const size_t nWp  = (size_t)EMB * EMB;               //   589,824
  const size_t per  = (size_t)BATCH * HEADS * SEQ * HD;

  char* ws = (char*)d_ws;
  __bf16* xb  = (__bf16*)ws;                   ws += nX  * sizeof(__bf16);
  __bf16* Wqt = (__bf16*)ws;                   ws += nWq * sizeof(__bf16);
  __bf16* Wpt = (__bf16*)ws;                   ws += nWp * sizeof(__bf16);
  __bf16* Qb  = (__bf16*)ws;                   ws += per * sizeof(__bf16);
  __bf16* Kb  = (__bf16*)ws;                   ws += per * sizeof(__bf16);
  __bf16* Vt  = (__bf16*)ws;                   ws += per * sizeof(__bf16);
  __bf16* Ob  = (__bf16*)ws;

  dim3 blk(256);
  cvt_bf16_kernel<<<dim3((unsigned)(nX / 4 / 256)), blk, 0, stream>>>(
      x, xb, (int)nX);
  transpose_cvt_kernel<<<dim3(QKV3 / 32, EMB / 32), blk, 0, stream>>>(
      qkv_w, Wqt, EMB, QKV3);
  transpose_cvt_kernel<<<dim3(EMB / 32, EMB / 32), blk, 0, stream>>>(
      proj_w, Wpt, EMB, EMB);

  qkv_kernel<<<dim3((BATCH * SEQ) / 128, QKV3 / 128), blk, 0, stream>>>(
      xb, Wqt, qkv_b, Qb, Kb, Vt);
  attn_kernel<<<dim3(SEQ / 32, BATCH), blk, 0, stream>>>(Qb, Kb, Vt, Ob);
  proj_kernel<<<dim3((BATCH * SEQ) / 128, EMB / 128), blk, 0, stream>>>(
      Ob, Wpt, proj_b, out);
}
